// SliceAttention_44667659878638
// MI455X (gfx1250) — compile-verified, matrix operand order checked by reading, not run on hardware
//
#include <hip/hip_runtime.h>

#define NN 4096
#define BM 128
#define BN 128
#define BK 32
#define LDT 40   // padded LDS row stride (bf16 elements): 40*2B=80B -> conflict-free

typedef __attribute__((ext_vector_type(16))) __bf16 v16bf;
typedef __attribute__((ext_vector_type(8)))  float  v8f;

struct __attribute__((aligned(16))) U4 { unsigned int x, y, z, w; };
union FragB16 { v16bf v; U4 u[2]; };

// round-to-nearest-even f32 -> bf16 bits
static __device__ __forceinline__ unsigned short f2bf_bits(float f) {
    unsigned int u = __float_as_uint(f);
    return (unsigned short)((u + 0x7FFFu + ((u >> 16) & 1u)) >> 16);
}
static __device__ __forceinline__ float bfbits2f(unsigned short s) {
    return __uint_as_float(((unsigned int)s) << 16);
}
static __device__ __forceinline__ unsigned int pack2(unsigned short lo, unsigned short hi) {
    return (unsigned int)lo | ((unsigned int)hi << 16);
}

// ---------------------------------------------------------------------------
// r[j] = sum_c outer[j, c]
// ---------------------------------------------------------------------------
__global__ __launch_bounds__(256)
void rowsum_kernel(const float* __restrict__ A, float* __restrict__ r) {
    __shared__ float red[256];
    const int row = blockIdx.x;
    const float4* p = (const float4*)(A + (size_t)row * NN);
    float s = 0.f;
    for (int j = threadIdx.x; j < NN / 4; j += 256) {
        float4 v = p[j];
        s += v.x + v.y + v.z + v.w;
    }
    red[threadIdx.x] = s;
    __syncthreads();
    for (int off = 128; off > 0; off >>= 1) {
        if (threadIdx.x < off) red[threadIdx.x] += red[threadIdx.x + off];
        __syncthreads();
    }
    if (threadIdx.x == 0) r[row] = red[0];
}

// ---------------------------------------------------------------------------
// S = sigmoid(A @ W^T + bias), A,W row-major [N,N].  Split-bf16 (hi/lo) WMMA:
// A*W ~= Ah*Wh + Ah*Wl + Al*Wh  with f32 accumulation -> ~fp32 accuracy.
// ---------------------------------------------------------------------------
__global__ __launch_bounds__(256)
void gemm_sigmoid_kernel(const float* __restrict__ A,
                         const float* __restrict__ W,
                         const float* __restrict__ bias,
                         float* __restrict__ S) {
    __shared__ unsigned short la_hi[BM * LDT], la_lo[BM * LDT];
    __shared__ unsigned short lb_hi[BN * LDT], lb_lo[BN * LDT];

    const int tid  = threadIdx.x;
    const int lane = tid & 31;
    const int wave = tid >> 5;
    const int wm   = wave >> 2;            // 0..1 -> 64 rows each
    const int wn   = wave & 3;             // 0..3 -> 32 cols each
    const int bm0  = blockIdx.y * BM;
    const int bn0  = blockIdx.x * BN;
    const int l15  = lane & 15;
    const int kb   = (lane >> 4) << 3;     // 0 or 8 (A/B 16-bit fragment K base)

    // accumulators initialized with the bias (bias[j] broadcasts over rows)
    v8f acc[4][2];
#pragma unroll
    for (int tn = 0; tn < 2; ++tn) {
        float bv = bias[bn0 + wn * 32 + tn * 16 + l15];
#pragma unroll
        for (int tm = 0; tm < 4; ++tm)
#pragma unroll
            for (int e = 0; e < 8; ++e) acc[tm][tn][e] = bv;
    }

    // staging: thread t converts row (t>>1), 16 K-values at (t&1)*16
    const int lrow = tid >> 1;
    const int lk   = (tid & 1) * 16;

    for (int k0 = 0; k0 < NN; k0 += BK) {
        __syncthreads();
        {   // stage A tile (rows of outer)
            const float4* src = (const float4*)(A + (size_t)(bm0 + lrow) * NN + k0 + lk);
            unsigned int* dh = (unsigned int*)(la_hi + lrow * LDT + lk);
            unsigned int* dl = (unsigned int*)(la_lo + lrow * LDT + lk);
            if (k0 + BK < NN) __builtin_prefetch((const float*)src + BK, 0, 0);
#pragma unroll
            for (int v = 0; v < 4; ++v) {
                float4 x = src[v];
                unsigned short h0 = f2bf_bits(x.x), h1 = f2bf_bits(x.y);
                unsigned short h2 = f2bf_bits(x.z), h3 = f2bf_bits(x.w);
                dh[2 * v + 0] = pack2(h0, h1);
                dh[2 * v + 1] = pack2(h2, h3);
                dl[2 * v + 0] = pack2(f2bf_bits(x.x - bfbits2f(h0)),
                                      f2bf_bits(x.y - bfbits2f(h1)));
                dl[2 * v + 1] = pack2(f2bf_bits(x.z - bfbits2f(h2)),
                                      f2bf_bits(x.w - bfbits2f(h3)));
            }
        }
        {   // stage B tile (rows of W; NT GEMM so W rows are K-contiguous)
            const float4* src = (const float4*)(W + (size_t)(bn0 + lrow) * NN + k0 + lk);
            unsigned int* dh = (unsigned int*)(lb_hi + lrow * LDT + lk);
            unsigned int* dl = (unsigned int*)(lb_lo + lrow * LDT + lk);
            if (k0 + BK < NN) __builtin_prefetch((const float*)src + BK, 0, 0);
#pragma unroll
            for (int v = 0; v < 4; ++v) {
                float4 x = src[v];
                unsigned short h0 = f2bf_bits(x.x), h1 = f2bf_bits(x.y);
                unsigned short h2 = f2bf_bits(x.z), h3 = f2bf_bits(x.w);
                dh[2 * v + 0] = pack2(h0, h1);
                dh[2 * v + 1] = pack2(h2, h3);
                dl[2 * v + 0] = pack2(f2bf_bits(x.x - bfbits2f(h0)),
                                      f2bf_bits(x.y - bfbits2f(h1)));
                dl[2 * v + 1] = pack2(f2bf_bits(x.z - bfbits2f(h2)),
                                      f2bf_bits(x.w - bfbits2f(h3)));
            }
        }
        __syncthreads();

        // load fragments (ISA 7.12.2: lane 0-15 holds K kb..kb+7 and kb+16..kb+23)
        FragB16 ah[4], al[4], bh[2], bl[2];
#pragma unroll
        for (int tm = 0; tm < 4; ++tm) {
            int base = (wm * 64 + tm * 16 + l15) * LDT + kb;
            ah[tm].u[0] = *(const U4*)(la_hi + base);
            ah[tm].u[1] = *(const U4*)(la_hi + base + 16);
            al[tm].u[0] = *(const U4*)(la_lo + base);
            al[tm].u[1] = *(const U4*)(la_lo + base + 16);
        }
#pragma unroll
        for (int tn = 0; tn < 2; ++tn) {
            int base = (wn * 32 + tn * 16 + l15) * LDT + kb;
            bh[tn].u[0] = *(const U4*)(lb_hi + base);
            bh[tn].u[1] = *(const U4*)(lb_hi + base + 16);
            bl[tn].u[0] = *(const U4*)(lb_lo + base);
            bl[tn].u[1] = *(const U4*)(lb_lo + base + 16);
        }

        // three passes interleaved over 8 accumulators -> no back-to-back D->C hazard
#pragma unroll
        for (int tm = 0; tm < 4; ++tm)
#pragma unroll
            for (int tn = 0; tn < 2; ++tn)
                acc[tm][tn] = __builtin_amdgcn_wmma_f32_16x16x32_bf16(
                    false, ah[tm].v, false, bh[tn].v, (short)0, acc[tm][tn], false, false);
#pragma unroll
        for (int tm = 0; tm < 4; ++tm)
#pragma unroll
            for (int tn = 0; tn < 2; ++tn)
                acc[tm][tn] = __builtin_amdgcn_wmma_f32_16x16x32_bf16(
                    false, ah[tm].v, false, bl[tn].v, (short)0, acc[tm][tn], false, false);
#pragma unroll
        for (int tm = 0; tm < 4; ++tm)
#pragma unroll
            for (int tn = 0; tn < 2; ++tn)
                acc[tm][tn] = __builtin_amdgcn_wmma_f32_16x16x32_bf16(
                    false, al[tm].v, false, bh[tn].v, (short)0, acc[tm][tn], false, false);
    }

    // epilogue: fused sigmoid, store f32
    // C/D layout: lane l, vgpr e -> M = e + 8*(l>=16), N = l&15
#pragma unroll
    for (int tm = 0; tm < 4; ++tm) {
        int row = bm0 + wm * 64 + tm * 16 + ((lane >> 4) << 3);
#pragma unroll
        for (int tn = 0; tn < 2; ++tn) {
            int col = bn0 + wn * 32 + tn * 16 + l15;
#pragma unroll
            for (int e = 0; e < 8; ++e) {
                float x = acc[tm][tn][e];
                S[(size_t)(row + e) * NN + col] = 1.0f / (1.0f + __expf(-x));
            }
        }
    }
}

// ---------------------------------------------------------------------------
// out[i] = sum_j exp(sq[i,j]*sk[j,i]) * r[j] / sum_j exp(sq[i,j]*sk[j,i])
// (scores in (0,1) -> softmax is safe without max subtraction)
// ---------------------------------------------------------------------------
__global__ __launch_bounds__(256)
void finalize_kernel(const float* __restrict__ SQ, const float* __restrict__ SK,
                     const float* __restrict__ r, float* __restrict__ out) {
    __shared__ float rn[256], rd[256];
    const int i = blockIdx.x;
    float num = 0.f, den = 0.f;
    for (int j = threadIdx.x; j < NN; j += 256) {
        float qv = SQ[(size_t)i * NN + j];
        float kv = SK[(size_t)j * NN + i];
        float e  = __expf(qv * kv);
        num += e * r[j];
        den += e;
    }
    rn[threadIdx.x] = num;
    rd[threadIdx.x] = den;
    __syncthreads();
    for (int off = 128; off > 0; off >>= 1) {
        if (threadIdx.x < off) {
            rn[threadIdx.x] += rn[threadIdx.x + off];
            rd[threadIdx.x] += rd[threadIdx.x + off];
        }
        __syncthreads();
    }
    if (threadIdx.x == 0) out[i] = rn[0] / rd[0];
}

extern "C" void kernel_launch(void* const* d_in, const int* in_sizes, int n_in,
                              void* d_out, int out_size, void* d_ws, size_t ws_size,
                              hipStream_t stream) {
    const float* outer  = (const float*)d_in[0];
    const float* Wslice = (const float*)d_in[1];
    const float* bslice = (const float*)d_in[2];
    const float* Wq     = (const float*)d_in[3];
    const float* bq     = (const float*)d_in[4];
    float* out = (float*)d_out;

    char*  ws = (char*)d_ws;
    float* SQ = (float*)ws;                                        // N*N f32
    float* SK = (float*)(ws + (size_t)NN * NN * sizeof(float));    // N*N f32
    float* r  = (float*)(ws + 2 * (size_t)NN * NN * sizeof(float)); // N f32

    rowsum_kernel<<<NN, 256, 0, stream>>>(outer, r);

    dim3 grid(NN / BN, NN / BM);
    gemm_sigmoid_kernel<<<grid, 256, 0, stream>>>(outer, Wq, bq, SQ);
    gemm_sigmoid_kernel<<<grid, 256, 0, stream>>>(outer, Wslice, bslice, SK);

    finalize_kernel<<<NN, 256, 0, stream>>>(SQ, SK, r, out);
}
